// RepWalk_81947976008110
// MI455X (gfx1250) — matrix operand order repaired and checked by reading, not compiled, not run on hardware
//
#include <hip/hip_runtime.h>
#include <hip/hip_bf16.h>
#include <math.h>

// ---------------- types for WMMA fragments ----------------
typedef __attribute__((ext_vector_type(16))) __bf16       v16bf;
typedef __attribute__((ext_vector_type(8)))  float        v8f;
typedef __attribute__((ext_vector_type(4)))  unsigned int u32x4;

union Frag { u32x4 q[2]; v16bf v; };

static __device__ __forceinline__ unsigned short f32_to_bf16(float f) {
  unsigned int u = __float_as_uint(f);
  unsigned int r = u + 0x7FFFu + ((u >> 16) & 1u);   // round-to-nearest-even
  return (unsigned short)(r >> 16);
}

// ---------------- problem constants ----------------
#define BSZ 32
#define SLN 256
#define NPC 320
#define WDI 768
#define KIN 896      // WD + 2*PD
#define FDI 512
#define NCV 1152     // (2+3+4)*128
#define NFT 128

// ---------------- kernel 1: repack fc1_W [896,512] f32 -> bf16 B-fragment layout ----------------
// layout: [kt][nt][lane][8 dwords]; lane: col = nt*16+(lane&15); K = kt*32 + (lane&16?16:0) + 2j(+1)
__global__ void repack_fc1_kernel(const float* __restrict__ W, unsigned int* __restrict__ Bp) {
  int id   = blockIdx.x * blockDim.x + threadIdx.x;  // 896*256 = 229376 = 28*32*32*8
  int j    = id & 7;
  int lane = (id >> 3) & 31;
  int nt   = (id >> 8) & 31;          // NT = 32
  int kt   = id >> 13;                // 28 k-tiles
  if (kt >= 28) return;
  int n  = nt * 16 + (lane & 15);
  int kb = kt * 32 + ((lane & 16) ? 16 : 0) + 2 * j;
  unsigned short lo = f32_to_bf16(W[(size_t)kb * FDI + n]);
  unsigned short hi = f32_to_bf16(W[(size_t)(kb + 1) * FDI + n]);
  Bp[id] = (unsigned int)lo | ((unsigned int)hi << 16);
}

// ---------------- kernel 2: repack conv weights (W_k2|W_k3|W_k4) -> [512,1152] bf16 B layout ----
__global__ void repack_conv_kernel(const float* __restrict__ W2, const float* __restrict__ W3,
                                   const float* __restrict__ W4, unsigned int* __restrict__ Bp) {
  int id   = blockIdx.x * blockDim.x + threadIdx.x;  // 1152*256 = 294912 = 16*72*32*8
  int j    = id & 7;
  int lane = (id >> 3) & 31;
  int rem  = id >> 8;
  int nt   = rem % 72;
  int kt   = rem / 72;
  if (kt >= 16) return;
  int c  = nt * 16 + (lane & 15);
  int kb = kt * 32 + ((lane & 16) ? 16 : 0) + 2 * j;
  const float* src; int k, m;
  if (c < 256)       { k = c >> 7;            m = c & 127;         src = W2; }
  else if (c < 640)  { int cc = c - 256; k = cc >> 7; m = cc & 127; src = W3; }
  else               { int cc = c - 640; k = cc >> 7; m = cc & 127; src = W4; }
  unsigned short lo = f32_to_bf16(src[((size_t)k * FDI + kb)     * NFT + m]);
  unsigned short hi = f32_to_bf16(src[((size_t)k * FDI + kb + 1) * NFT + m]);
  Bp[((size_t)(kt * 72 + nt) * 32 + lane) * 8 + j] = (unsigned int)lo | ((unsigned int)hi << 16);
}

// ---------------- kernel 3: wordpiece average + embeddings -> node_in bf16 [8192,896] ----------
__global__ void build_node_in_kernel(const float* __restrict__ bert, const int* __restrict__ wpi,
                                     const int* __restrict__ pos, const int* __restrict__ path,
                                     const float* __restrict__ pos_emb, const float* __restrict__ pd_emb,
                                     unsigned short* __restrict__ node_in) {
  int row = blockIdx.x;             // b*SL + t
  int b   = row >> 8;
  const int* wi = wpi + row * 4;
  int i0 = wi[0], i1 = wi[1], i2 = wi[2], i3 = wi[3];
  int cnt = (i0 != 0) + (i1 != 0) + (i2 != 0) + (i3 != 0);
  float inv = 1.0f / (float)(cnt > 0 ? cnt : 1);
  int pv = pos[row];
  const int* pr = path + row * 8;
  int pd = 0;
  #pragma unroll
  for (int p = 0; p < 8; p++) pd += (pr[p] != 0);
  const float* base = bert + (size_t)b * NPC * WDI;
  unsigned short* orow = node_in + (size_t)row * KIN;
  for (int f = threadIdx.x; f < KIN; f += blockDim.x) {
    float v;
    if (f < WDI) {
      float s = 0.f;
      if (i0) s += base[(size_t)(i0 - 1) * WDI + f];
      if (i1) s += base[(size_t)(i1 - 1) * WDI + f];
      if (i2) s += base[(size_t)(i2 - 1) * WDI + f];
      if (i3) s += base[(size_t)(i3 - 1) * WDI + f];
      v = s * inv;
    } else if (f < WDI + 64) {
      v = pos_emb[pv * 64 + (f - WDI)];
    } else {
      v = pd_emb[pd * 64 + (f - WDI - 64)];
    }
    orow[f] = f32_to_bf16(v);
  }
}

// ---------------- kernel 4: WMMA GEMM  C[M,N] = act(A_bf16[M,K] @ Bpack + bias) -----------------
// one wave per block; wave computes a 32x64 tile (2 A frags x 4 B frags -> 8 accumulators).
// single-buffered fragments (~128 VGPRs -> 8 waves/SIMD); latency hidden by wave parallelism:
// per k-tile: one clause of 12 b128 loads, one wait, 8 back-to-back WMMAs.
template <int K, int RELU>
__global__ void wmma_gemm_kernel(const unsigned short* __restrict__ A,
                                 const unsigned int* __restrict__ Bp,
                                 const float* __restrict__ bias, float* __restrict__ C,
                                 int N) {
  constexpr int KT = K >> 5;                   // 28 or 16
  int lane  = threadIdx.x;
  int mtile = blockIdx.x;                      // 32-row tiles
  int ngrp  = blockIdx.y;                      // 64-col groups
  const unsigned short* arow0 = A + (size_t)(mtile * 32 + (lane & 15)) * K;
  const unsigned short* arow1 = arow0 + (size_t)16 * K;
  int koff  = (lane & 16) ? 16 : 0;            // contiguous k order (B packed alike): hi lanes hold K+16..K+31
  int NT16  = N >> 4;
  const unsigned int* bcol = Bp + (size_t)ngrp * 4 * 256 + lane * 8;

  v8f z = {};
  v8f acc[8] = {z, z, z, z, z, z, z, z};

  #pragma unroll 1
  for (int kt = 0; kt < KT; kt++) {
    int kb = kt * 32 + koff;
    const unsigned int* bb = bcol + (size_t)kt * NT16 * 256;
    // ---- all fragment loads first (one clause, one wait) ----
    Frag a0, a1, b0, b1, b2, b3;
    a0.q[0] = *(const u32x4*)(arow0 + kb);
    a0.q[1] = *(const u32x4*)(arow0 + kb + 8);
    a1.q[0] = *(const u32x4*)(arow1 + kb);
    a1.q[1] = *(const u32x4*)(arow1 + kb + 8);
    b0.q[0] = *(const u32x4*)(bb);
    b0.q[1] = *(const u32x4*)(bb + 4);
    b1.q[0] = *(const u32x4*)(bb + 256);
    b1.q[1] = *(const u32x4*)(bb + 260);
    b2.q[0] = *(const u32x4*)(bb + 512);
    b2.q[1] = *(const u32x4*)(bb + 516);
    b3.q[0] = *(const u32x4*)(bb + 768);
    b3.q[1] = *(const u32x4*)(bb + 772);
    if (kt + 1 < KT) __builtin_prefetch(arow0 + kb + 32, 0, 3);
    // ---- 8 back-to-back WMMAs ----
    acc[0] = __builtin_amdgcn_wmma_f32_16x16x32_bf16(false, a0.v, false, b0.v, (short)0, acc[0], false, false);
    acc[1] = __builtin_amdgcn_wmma_f32_16x16x32_bf16(false, a1.v, false, b0.v, (short)0, acc[1], false, false);
    acc[2] = __builtin_amdgcn_wmma_f32_16x16x32_bf16(false, a0.v, false, b1.v, (short)0, acc[2], false, false);
    acc[3] = __builtin_amdgcn_wmma_f32_16x16x32_bf16(false, a1.v, false, b1.v, (short)0, acc[3], false, false);
    acc[4] = __builtin_amdgcn_wmma_f32_16x16x32_bf16(false, a0.v, false, b2.v, (short)0, acc[4], false, false);
    acc[5] = __builtin_amdgcn_wmma_f32_16x16x32_bf16(false, a1.v, false, b2.v, (short)0, acc[5], false, false);
    acc[6] = __builtin_amdgcn_wmma_f32_16x16x32_bf16(false, a0.v, false, b3.v, (short)0, acc[6], false, false);
    acc[7] = __builtin_amdgcn_wmma_f32_16x16x32_bf16(false, a1.v, false, b3.v, (short)0, acc[7], false, false);
  }

  // D layout: vgpr j, lanes 0-15 -> (M=j, N=lane); lanes 16-31 -> (M=j+8, N=lane-16)
  int mbase = mtile * 32 + ((lane & 16) ? 8 : 0);
  int colb  = ngrp * 64 + (lane & 15);
  #pragma unroll
  for (int t = 0; t < 4; t++) {
    int col  = colb + t * 16;
    float bv = bias ? bias[col] : 0.f;
    #pragma unroll
    for (int r = 0; r < 2; r++) {
      int mrow = mbase + r * 16;
      #pragma unroll
      for (int j = 0; j < 8; j++) {
        float v = acc[t * 2 + r][j] + bv;
        if (RELU) v = fmaxf(v, 0.f);
        C[(size_t)(mrow + j) * N + col] = v;
      }
    }
  }
}

// ---------------- kernel 5: f32 -> bf16 copy of node for second GEMM ----------------------------
__global__ void cvt_bf16_kernel(const float* __restrict__ src, unsigned short* __restrict__ dst, int n) {
  int i = blockIdx.x * blockDim.x + threadIdx.x;
  if (i < n) dst[i] = f32_to_bf16(src[i]);
}

// ---------------- kernel 6: bilinear precompute: M[64,1024] = deprel_emb @ bilin_W^T, v1 = W1@ext
__global__ void bilin_pre_kernel(const float* __restrict__ dep_emb, const float* __restrict__ bw,
                                 const float* __restrict__ ext_rel,
                                 float* __restrict__ Mmat, float* __restrict__ v1) {
  int d = blockIdx.x * blockDim.x + threadIdx.x;   // 1024
  if (d >= 1024) return;
  const float* wrow = bw + (size_t)d * 64;
  for (int r = 0; r < 64; r++) {
    const float* er = dep_emb + r * 64;
    float s = 0.f;
    for (int j = 0; j < 64; j++) s += er[j] * wrow[j];
    Mmat[r * 1024 + d] = s;
  }
  if (d < FDI) {
    float s = 0.f;
    for (int j = 0; j < 64; j++) s += wrow[j] * ext_rel[j];
    v1[d] = s;
  }
}

// ---------------- kernel 7: per-row edge scores (one wave per token) ----------------------------
__global__ void scores_kernel(const float* __restrict__ node, const int* __restrict__ aspect_head,
                              const int* __restrict__ deprel, const float* __restrict__ Mmat,
                              const float* __restrict__ v1, const float* __restrict__ bilin_b,
                              float* __restrict__ score1, float* __restrict__ score2) {
  int gid  = blockIdx.x * blockDim.x + threadIdx.x;
  int wave = gid >> 5;                // exactly 8192 waves launched
  int lane = threadIdx.x & 31;
  int b    = wave >> 8;
  const float* nrow = node + (size_t)wave * FDI;
  int h = aspect_head[wave];
  const float* drow = (h > 0) ? node + (size_t)(b * SLN + h - 1) * FDI : nullptr;
  int dr = deprel[wave];
  const float* m1 = Mmat + (size_t)dr * 1024;
  const float* m2 = m1 + FDI;
  float s1 = 0.f, s2 = 0.f;
  for (int d = lane; d < FDI; d += 32) {
    float nd = nrow[d];
    float dd = drow ? drow[d] : 0.f;
    s1 += dd * m1[d] + nd * m2[d];
    s2 += nd * v1[d];
  }
  #pragma unroll
  for (int off = 16; off > 0; off >>= 1) {
    s1 += __shfl_xor(s1, off, 32);
    s2 += __shfl_xor(s2, off, 32);
  }
  if (lane == 0) {
    float bb = bilin_b[0];
    score1[wave] = 1.0f / (1.0f + expf(-(s1 + bb)));
    score2[wave] = 1.0f / (1.0f + expf(-(s2 + bb)));
  }
}

// ---------------- kernel 8: node weight via path product + weight_norm reduction ----------------
__global__ void nodeweight_kernel(const int* __restrict__ path, const int* __restrict__ tmask,
                                  const int* __restrict__ amask, const float* __restrict__ score1,
                                  const float* __restrict__ score2, float* __restrict__ nw,
                                  float* __restrict__ wn_out) {
  int b = blockIdx.x, t = threadIdx.x;
  int row = b * SLN + t;
  const int* pr = path + row * 8;
  float prod = 1.f;
  #pragma unroll
  for (int p = 0; p < 8; p++) {
    int jidx = pr[p];
    float es = 1.f;
    if (jidx >= 1 && jidx <= SLN)      es = score1[b * SLN + jidx - 1];
    else if (jidx > SLN)               es = score2[b * SLN + jidx - 1 - SLN];
    prod *= es;
  }
  if (!((tmask[row] != 0) && (amask[row] == 0))) prod = 0.f;
  nw[row] = prod;
  __shared__ float red[SLN];
  red[t] = prod;
  __syncthreads();
  for (int s = SLN / 2; s > 0; s >>= 1) {
    if (t < s) red[t] += red[t + s];
    __syncthreads();
  }
  if (t == 0) wn_out[b] = red[0];
}

// ---------------- kernel 9: windowed scale + bias + max pool ------------------------------------
__global__ void maxpool_kernel(const float* __restrict__ conv, const float* __restrict__ nw,
                               const float* __restrict__ b2, const float* __restrict__ b3,
                               const float* __restrict__ b4, float* __restrict__ pooled) {
  int b   = blockIdx.x;
  int kwi = blockIdx.y;          // 0,1,2 -> kw 2,3,4
  int m   = threadIdx.x;         // 128 feats
  int kw  = kwi + 2;
  int base = (kwi == 0) ? 0 : (kwi == 1 ? 256 : 640);
  const float* bias = (kwi == 0) ? b2 : (kwi == 1 ? b3 : b4);
  int cn = SLN - kw + 1;
  const float* cb  = conv + (size_t)b * SLN * NCV;
  const float* nwb = nw + b * SLN;
  float invkw = 1.0f / (float)kw;
  float mx = -3.4e38f;
  for (int n = 0; n < cn; n++) {
    float s = 0.f, w = 0.f;
    for (int k = 0; k < kw; k++) {
      s += cb[(size_t)(n + k) * NCV + base + k * NFT + m];
      w += nwb[n + k];
    }
    float val = (w * invkw) * s + bias[m];
    mx = fmaxf(mx, val);
  }
  pooled[(b * 3 + kwi) * NFT + m] = mx;
}

// ---------------- kernel 10: final FC [B,384] @ [384,3] -----------------------------------------
__global__ void fcout_kernel(const float* __restrict__ pooled, const float* __restrict__ W,
                             const float* __restrict__ bias, float* __restrict__ out) {
  int id = blockIdx.x * blockDim.x + threadIdx.x;
  if (id >= BSZ * 3) return;
  int b = id / 3, o = id % 3;
  float s = bias[o];
  const float* pr = pooled + b * 384;
  for (int j = 0; j < 384; j++) s += pr[j] * W[j * 3 + o];
  out[id] = s;
}

// ---------------- host orchestration ------------------------------------------------------------
extern "C" void kernel_launch(void* const* d_in, const int* in_sizes, int n_in,
                              void* d_out, int out_size, void* d_ws, size_t ws_size,
                              hipStream_t stream) {
  (void)in_sizes; (void)n_in; (void)out_size; (void)ws_size;
  const float* bert      = (const float*)d_in[0];
  const int*   wpi       = (const int*)  d_in[1];
  const int*   pos       = (const int*)  d_in[2];
  const int*   deprel    = (const int*)  d_in[3];
  const int*   path      = (const int*)  d_in[4];
  const int*   ahead     = (const int*)  d_in[5];
  const int*   tmask     = (const int*)  d_in[6];
  const int*   amask     = (const int*)  d_in[7];
  const float* pos_emb   = (const float*)d_in[8];
  const float* dep_emb   = (const float*)d_in[9];
  const float* pd_emb    = (const float*)d_in[10];
  const float* ext_rel   = (const float*)d_in[11];
  const float* fc1_W     = (const float*)d_in[12];
  const float* fc1_b     = (const float*)d_in[13];
  const float* bilin_W   = (const float*)d_in[14];
  const float* bilin_b   = (const float*)d_in[15];
  const float* W_k2      = (const float*)d_in[16];
  const float* b_k2      = (const float*)d_in[17];
  const float* W_k3      = (const float*)d_in[18];
  const float* b_k3      = (const float*)d_in[19];
  const float* W_k4      = (const float*)d_in[20];
  const float* b_k4      = (const float*)d_in[21];
  const float* fco_W     = (const float*)d_in[22];
  const float* fco_b     = (const float*)d_in[23];

  char* ws = (char*)d_ws;
  // all sizes are multiples of 256 bytes
  unsigned int*   bpack1  = (unsigned int*)  (ws + 0);            //  917504 B
  unsigned int*   bpack2  = (unsigned int*)  (ws + 917504);       // 1179648 B
  unsigned short* node_in = (unsigned short*)(ws + 2097152);      // 14680064 B
  float*          node    = (float*)         (ws + 16777216);     // 16777216 B
  unsigned short* node_bf = (unsigned short*)(ws + 33554432);     //  8388608 B
  float*          Mmat    = (float*)         (ws + 41943040);     //   262144 B
  float*          v1      = (float*)         (ws + 42205184);     //     2048 B
  float*          score1  = (float*)         (ws + 42207232);     //    32768 B
  float*          score2  = (float*)         (ws + 42240000);     //    32768 B
  float*          nw      = (float*)         (ws + 42272768);     //    32768 B
  float*          pooled  = (float*)         (ws + 42305536);     //    49152 B
  float*          conv    = (float*)         (ws + 42354688);     // 37748736 B -> 80103424 total

  float* out_pred = (float*)d_out;        // [32,3]
  float* out_wn   = (float*)d_out + 96;   // [32]

  // 1) weight repacks (bf16 B-fragment layout)
  repack_fc1_kernel<<<896, 256, 0, stream>>>(fc1_W, bpack1);
  repack_conv_kernel<<<1152, 256, 0, stream>>>(W_k2, W_k3, W_k4, bpack2);
  // 2) build node_in (wordpiece avg + pos/pathdepth embeddings) as bf16
  build_node_in_kernel<<<BSZ * SLN, 128, 0, stream>>>(bert, wpi, pos, path, pos_emb, pd_emb, node_in);
  // 3) GEMM1: node = relu(node_in @ fc1_W + b)   [8192,896]x[896,512], 32x64 tiles
  wmma_gemm_kernel<KIN, 1><<<dim3(256, 8), 32, 0, stream>>>(node_in, bpack1, fc1_b, node, FDI);
  // 4) bf16 copy of node for GEMM2
  cvt_bf16_kernel<<<16384, 256, 0, stream>>>(node, node_bf, BSZ * SLN * FDI);
  // 5) bilinear precompute (64 labels -> collapses einsum)
  bilin_pre_kernel<<<4, 256, 0, stream>>>(dep_emb, bilin_W, ext_rel, Mmat, v1);
  // 6) per-token edge scores
  scores_kernel<<<1024, 256, 0, stream>>>(node, ahead, deprel, Mmat, v1, bilin_b, score1, score2);
  // 7) node weights + weight_norm output
  nodeweight_kernel<<<BSZ, SLN, 0, stream>>>(path, tmask, amask, score1, score2, nw, out_wn);
  // 8) GEMM2: conv = node @ [W2|W3|W4]   [8192,512]x[512,1152], 32x64 tiles
  wmma_gemm_kernel<FDI, 0><<<dim3(256, 18), 32, 0, stream>>>(node_bf, bpack2, nullptr, conv, NCV);
  // 9) windowed scale+bias+maxpool
  maxpool_kernel<<<dim3(BSZ, 3), NFT, 0, stream>>>(conv, nw, b_k2, b_k3, b_k4, pooled);
  // 10) final FC -> predicts
  fcout_kernel<<<1, 96, 0, stream>>>(pooled, fco_W, fco_b, out_pred);
}
